// GTransformerPretrainImproved_6141803233663
// MI455X (gfx1250) — matrix operand order repaired and checked by reading, not compiled, not run on hardware
//
#include <hip/hip_runtime.h>

// Problem constants (match reference)
#define B_   32
#define N_   2048
#define L_   32768
#define TILES_PER_BATCH (L_/16)        // 2048
#define NTILES (B_*TILES_PER_BATCH)    // 65536
#define K2_BLOCKS 512
#define K2_WAVES  (K2_BLOCKS*8)        // 4096 waves, 16 tiles each

typedef __attribute__((ext_vector_type(16))) _Float16 v16h;
typedef __attribute__((ext_vector_type(8)))  float    v8f;

#if defined(__has_builtin)
#  if __has_builtin(__builtin_amdgcn_sched_group_barrier)
#    define SCHED_GROUP(mask, size, sync) __builtin_amdgcn_sched_group_barrier(mask, size, sync)
#  else
#    define SCHED_GROUP(mask, size, sync)
#  endif
#else
#  define SCHED_GROUP(mask, size, sync)
#endif

union HPack { _Float16 h; unsigned short u; };
__device__ __forceinline__ unsigned short f2h(float f) {
  HPack p; p.h = (_Float16)f; return p.u;
}

union AOp { uint4 q[2]; v16h v; };   // 32 bytes = 8 VGPRs per lane

// ---------------------------------------------------------------------------
// Kernel 1: node embedding (B*N,4) @ (4,64) + b -> f16, packed 2/dword
// ---------------------------------------------------------------------------
__global__ void k_embed(const float* __restrict__ pred,
                        const float* __restrict__ We,
                        const float* __restrict__ be,
                        unsigned int* __restrict__ nf) {
  int tid  = blockIdx.x * 256 + threadIdx.x;   // 0 .. B*N*32-1
  int node = tid >> 5;
  int pair = tid & 31;
  int d0   = pair * 2;
  const float* pr = pred + (size_t)node * 4;
  float p0 = pr[0], p1 = pr[1], p2 = pr[2], p3 = pr[3];
  float a = be[d0]     + p0*We[d0]     + p1*We[64+d0]     + p2*We[128+d0]     + p3*We[192+d0];
  float b = be[d0+1]   + p0*We[d0+1]   + p1*We[64+d0+1]   + p2*We[128+d0+1]   + p3*We[192+d0+1];
  nf[(size_t)node * 32 + pair] = (unsigned)f2h(a) | ((unsigned)f2h(b) << 16);
}

// ---------------------------------------------------------------------------
// Kernel 2: pack W1 (128,64) fp32 into per-lane WMMA B-operand layout, f16.
// dword index = combo*256 + lane*8 + v,  combo = kt*4+nt
//   k = 32*kt + 16*(lane>>4) + 2*v (+pair),  n = 16*nt + (lane&15)
// ---------------------------------------------------------------------------
__global__ void k_packw1(const float* __restrict__ W1,
                         unsigned int* __restrict__ w1p) {
  int idx   = blockIdx.x * 256 + threadIdx.x;  // 0..4095
  int v     = idx & 7;
  int lane  = (idx >> 3) & 31;
  int combo = idx >> 8;                        // 0..15
  int kt = combo >> 2, nt = combo & 3;
  int k = 32*kt + 16*(lane >> 4) + 2*v;
  int n = 16*nt + (lane & 15);
  unsigned lo = f2h(W1[k * 64 + n]);
  unsigned hi = f2h(W1[(k + 1) * 64 + n]);
  w1p[combo * 256 + lane * 8 + v] = lo | (hi << 16);
}

// ---------------------------------------------------------------------------
// Kernel 3: edge MLP. One wave per 16-edge tile (grid-stride, 16 tiles/wave).
// B matrix held in 128 VGPRs for the whole loop. 16 WMMAs per tile.
// Scheduling: all 8 A-gather VMEM reads issued up front, then the 16 WMMAs,
// so later gathers overlap matrix math instead of 4x full loadcnt stalls.
// ---------------------------------------------------------------------------
__global__ void __launch_bounds__(256) k_edge(
    const unsigned short* __restrict__ nf,   // (B*N, 64) f16 node features
    const unsigned int*  __restrict__ w1p,   // packed W1 B-operands
    const float* __restrict__ b1,            // (64,)
    const float* __restrict__ W2,            // (64,4)
    const float* __restrict__ b2,            // (4,)
    const int*   __restrict__ esrc,          // (B,L)
    const int*   __restrict__ edst,          // (B,L)
    const float* __restrict__ lp,            // (B,L,4) line_param
    float* __restrict__ out)                 // (B,L,4)
{
  const int lane = threadIdx.x & 31;
  const int w    = threadIdx.x >> 5;
  const int gw   = blockIdx.x * 8 + w;
  const int half = lane >> 4;
  const int nl   = lane & 15;

  // --- register-resident B operands: 16 combos x 8 VGPRs = 128 VGPRs ---
  AOp bm[16];
#pragma unroll
  for (int c = 0; c < 16; ++c) {
    const uint4* p = reinterpret_cast<const uint4*>(w1p + c * 256 + lane * 8);
    bm[c].q[0] = p[0];
    bm[c].q[1] = p[1];
  }

  // per-lane constants: bias (C-layout: depends only on n = 16*nt + nl)
  float b1v[4], w2a[4], w2b[4];
#pragma unroll
  for (int nt = 0; nt < 4; ++nt) {
    b1v[nt] = b1[nt * 16 + nl];
    w2a[nt] = W2[(nt * 16 + nl) * 4 + 2];
    w2b[nt] = W2[(nt * 16 + nl) * 4 + 3];
  }
  const float b2c = b2[2], b2d = b2[3];

  for (int t = gw; t < NTILES; t += K2_WAVES) {
    const int b    = t >> 11;                       // tile / 2048
    const int base = (t & (TILES_PER_BATCH - 1)) * 16;
    const int eidx = b * L_ + base + nl;            // this lane's edge (m = nl)

    const int s = esrc[eidx];
    const int d = edst[eidx];
    const unsigned short* rf = nf + ((size_t)b * N_ + s) * 64;
    const unsigned short* rj = nf + ((size_t)b * N_ + d) * 64;

    // --- A operands in contiguous k order (as packed W1): lane<16 -> K 0..15, lane>=16 -> K 16..31 ---
    AOp a[4];
#pragma unroll
    for (int kt = 0; kt < 2; ++kt) {
      const int e0 = 32 * kt + 16 * half;
      a[kt].q[0]     = *reinterpret_cast<const uint4*>(rf + e0);
      a[kt].q[1]     = *reinterpret_cast<const uint4*>(rf + e0 + 8);
      a[kt + 2].q[0] = *reinterpret_cast<const uint4*>(rj + e0);
      a[kt + 2].q[1] = *reinterpret_cast<const uint4*>(rj + e0 + 8);
    }

    // --- 16 edges x 64 outputs: 4 N-tiles x 4 K-steps of wmma f16 ---
    v8f acc[4];
#pragma unroll
    for (int nt = 0; nt < 4; ++nt) {
      v8f c;
#pragma unroll
      for (int r = 0; r < 8; ++r) c[r] = b1v[nt];   // fused +b1
#pragma unroll
      for (int kt = 0; kt < 4; ++kt)
        c = __builtin_amdgcn_wmma_f32_16x16x32_f16(
                false, a[kt].v, false, bm[kt * 4 + nt].v,
                (short)0, c, false, false);
      acc[nt] = c;
    }

    // Force pipeline shape: 10 VMEM reads (A gathers + indices) first,
    // then the 16 WMMAs (partial loadcnt waits overlap loads with math).
    SCHED_GROUP(0x020, 10, 0);  // VMEM reads
    SCHED_GROUP(0x008, 16, 0);  // MFMA/WMMA

    // --- ReLU + 64->2 projection (cols 2,3 only), cross-lane reduce ---
    float p2 = 0.f, p3 = 0.f;
#pragma unroll
    for (int r = 0; r < 8; ++r) {
      float s2 = 0.f, s3 = 0.f;
#pragma unroll
      for (int nt = 0; nt < 4; ++nt) {
        float h = acc[nt][r];
        h = h > 0.f ? h : 0.f;
        s2 += h * w2a[nt];
        s3 += h * w2b[nt];
      }
      // butterfly within each 16-lane half: sum over n = 0..63
      s2 += __shfl_xor(s2, 8, 32);  s3 += __shfl_xor(s3, 8, 32);
      s2 += __shfl_xor(s2, 4, 32);  s3 += __shfl_xor(s3, 4, 32);
      s2 += __shfl_xor(s2, 2, 32);  s3 += __shfl_xor(s3, 2, 32);
      s2 += __shfl_xor(s2, 1, 32);  s3 += __shfl_xor(s3, 1, 32);
      // bring the other half's value (edge m = r+8) across
      float x2 = __shfl_xor(s2, 16, 32);
      float x3 = __shfl_xor(s3, 16, 32);
      // lane nl (edge m = nl) picks its value: m<8 from own half, else other
      if ((nl & 7) == r) {
        bool low = nl < 8;
        p2 = low ? s2 : x2;
        p3 = low ? s3 : x3;
      }
    }

    // --- coalesced full-row output: [lp0, lp1, pred2, pred3] ---
    if (lane < 16) {
      float2 l01 = *reinterpret_cast<const float2*>(lp + (size_t)eidx * 4);
      float4 o4;
      o4.x = l01.x;
      o4.y = l01.y;
      o4.z = p2 + b2c;
      o4.w = p3 + b2d;
      *reinterpret_cast<float4*>(out + (size_t)eidx * 4) = o4;
    }
  }
}

// ---------------------------------------------------------------------------
extern "C" void kernel_launch(void* const* d_in, const int* in_sizes, int n_in,
                              void* d_out, int out_size, void* d_ws, size_t ws_size,
                              hipStream_t stream) {
  const float* pred_node = (const float*)d_in[0];
  const float* line_param= (const float*)d_in[1];
  const float* W_embed   = (const float*)d_in[2];
  const float* b_embed   = (const float*)d_in[3];
  const float* W1        = (const float*)d_in[4];
  const float* b1        = (const float*)d_in[5];
  const float* W2        = (const float*)d_in[6];
  const float* b2        = (const float*)d_in[7];
  const int*   edge_src  = (const int*)d_in[8];
  const int*   edge_dst  = (const int*)d_in[9];
  float* out = (float*)d_out;

  // workspace layout: [0, 8MB) node features f16; [8MB, +16KB) packed W1
  unsigned int* nf_u32 = (unsigned int*)d_ws;
  unsigned int* w1p    = nf_u32 + (size_t)B_ * N_ * 32;   // 2M dwords = 8 MB

  // 1) node embedding: B*N*32 threads (2 outputs each)
  k_embed<<<(B_ * N_ * 32) / 256, 256, 0, stream>>>(pred_node, W_embed, b_embed, nf_u32);

  // 2) pack W1 into WMMA B-operand layout: 4096 threads
  k_packw1<<<16, 256, 0, stream>>>(W1, w1p);

  // 3) edge MLP with WMMA
  k_edge<<<K2_BLOCKS, 256, 0, stream>>>(
      (const unsigned short*)nf_u32, w1p, b1, W2, b2,
      edge_src, edge_dst, line_param, out);
}
